// SPAttnHead_44470091383216
// MI455X (gfx1250) — compile-verified, matrix operand order checked by reading, not run on hardware
//
#include <hip/hip_runtime.h>
#include <hip/hip_bf16.h>

#define IN_CH 256
#define OUT_CH 128
#define NN 6144

typedef __attribute__((ext_vector_type(16))) _Float16 v16h;
typedef __attribute__((ext_vector_type(8)))  float    v8f;

// ---------------------------------------------------------------------------
// Kernel 1: f[o][n] = sum_i W[o][i] * x[i][n]   (f: [128, 6144] f32)
// ---------------------------------------------------------------------------
__global__ __launch_bounds__(256) void fts_kernel(const float* __restrict__ W,
                                                  const float* __restrict__ x,
                                                  float* __restrict__ f) {
    const int n = blockIdx.x * 256 + threadIdx.x;
    const int o = blockIdx.y;
    const float* wr = W + o * IN_CH;
    float acc = 0.f;
#pragma unroll 8
    for (int i = 0; i < IN_CH; ++i)
        acc = fmaf(wr[i], x[(size_t)i * NN + n], acc);
    f[(size_t)o * NN + n] = acc;
}

// ---------------------------------------------------------------------------
// Kernel 2: f1[n] = w1 . f[:,n] + b1 ;  f2[n] = w2 . f[:,n] + b2
// ---------------------------------------------------------------------------
__global__ __launch_bounds__(256) void f12_kernel(const float* __restrict__ f,
                                                  const float* __restrict__ w1,
                                                  const float* __restrict__ b1,
                                                  const float* __restrict__ w2,
                                                  const float* __restrict__ b2,
                                                  float* __restrict__ f1,
                                                  float* __restrict__ f2) {
    const int n = blockIdx.x * 256 + threadIdx.x;
    float s1 = b1[0], s2 = b2[0];
#pragma unroll 8
    for (int c = 0; c < OUT_CH; ++c) {
        const float v = f[(size_t)c * NN + n];
        s1 = fmaf(w1[c], v, s1);
        s2 = fmaf(w2[c], v, s2);
    }
    f1[n] = s1;
    f2[n] = s2;
}

// ---------------------------------------------------------------------------
// Kernel 3: pre-swizzle B = f^T into the WMMA 16-bit B-operand lane layout.
// Bp[kt][cb][lane][h] = (f16) f[cb*16 + (lane&15)][kt*32 + Kmap(lane,h)]
// 16-bit A/B layout: halves 0..7 -> K = 8*(lane>>4) + h,
//                    halves 8..15 -> K = 16 + 8*(lane>>4) + (h-8)
// ---------------------------------------------------------------------------
__device__ __forceinline__ int kmap(int lhalf, int h) {
    return (h < 8) ? (lhalf * 8 + h) : (16 + lhalf * 8 + (h - 8));
}

__global__ __launch_bounds__(256) void pack_kernel(const float* __restrict__ f,
                                                   _Float16* __restrict__ Bp) {
    const int idx  = blockIdx.x * 256 + threadIdx.x; // over 192*8*32*16
    const int h    = idx & 15;
    const int lane = (idx >> 4) & 31;
    const int cb   = (idx >> 9) & 7;
    const int kt   = idx >> 12;
    const int c    = cb * 16 + (lane & 15);
    const int k    = kt * 32 + kmap(lane >> 4, h);
    Bp[idx] = (_Float16)f[(size_t)c * NN + k];
}

// ---------------------------------------------------------------------------
// Kernel 4: single-pass fused sparse-softmax attention + GEMM.
// One wave32 per 16 output rows.  Per K-step:
//   - each lane loads its own row's two 8-float runs of adj (4x float4)
//   - builds UNNORMALIZED exp(leaky(f1+f2)) A-tile in f16, accumulating the
//     f32 row-sum alongside (normalization is a per-row linear scale and is
//     applied to the WMMA accumulators in the epilogue -> adj read ONCE)
//   - 8x v_wmma_f32_16x16x32_f16 against pre-swizzled B tiles
// ---------------------------------------------------------------------------
__global__ __launch_bounds__(32) void attn_kernel(const float* __restrict__ adj,
                                                  const float* __restrict__ f1,
                                                  const float* __restrict__ f2,
                                                  const _Float16* __restrict__ Bp,
                                                  const float* __restrict__ vbias,
                                                  float* __restrict__ out) {
    const int lane  = threadIdx.x;          // 0..31, one full wave
    const int row0  = blockIdx.x * 16;
    const int mrow  = lane & 15;            // this lane's A-row / C-column
    const int lhalf = lane >> 4;

    const float f1m = f1[row0 + mrow];

    const float4* arow = (const float4*)(adj + (size_t)(row0 + mrow) * NN);
    const float4* f2v  = (const float4*)f2;

    v8f acc[8];
#pragma unroll
    for (int cb = 0; cb < 8; ++cb)
        acc[cb] = v8f{0.f, 0.f, 0.f, 0.f, 0.f, 0.f, 0.f, 0.f};

    float psum = 0.f;   // this lane's partial of row mrow's softmax denom

    for (int kt = 0; kt < NN / 32; ++kt) {
        // A-operand K positions for this lane: two runs of 8 floats
        const int base = kt * 8 + lhalf * 2;     // float4 index into row
        float4 a4[4], t4[4];
        a4[0] = arow[base];     a4[1] = arow[base + 1];
        a4[2] = arow[base + 4]; a4[3] = arow[base + 5];
        t4[0] = f2v[base];      t4[1] = f2v[base + 1];
        t4[2] = f2v[base + 4];  t4[3] = f2v[base + 5];

        v16h a;
#pragma unroll
        for (int q = 0; q < 4; ++q) {
            const float* ap = reinterpret_cast<const float*>(&a4[q]);
            const float* tp = reinterpret_cast<const float*>(&t4[q]);
#pragma unroll
            for (int e = 0; e < 4; ++e) {
                const float t  = f1m + tp[e];
                const float s  = (t < 0.f) ? 0.2f * t : t;   // leaky_relu
                const float ex = __expf(s);
                const float cf = ((ap[e] != 0.f) && (t != 0.f)) ? ex : 0.f;
                psum += cf;
                a[q * 4 + e] = (_Float16)cf;
            }
        }

        // 8 pre-swizzled B tiles: one contiguous 32B v16h load per lane/tile
        const v16h* bp = (const v16h*)(Bp + (size_t)kt * 8 * 32 * 16);
#pragma unroll
        for (int cb = 0; cb < 8; ++cb) {
            const v16h b = bp[cb * 32 + lane];
            acc[cb] = __builtin_amdgcn_wmma_f32_16x16x32_f16(
                false, a, false, b, (short)0, acc[cb], false, false);
        }
    }

    // Row softmax denominator: lanes l and l+16 cover disjoint K halves
    const float rsum = psum + __shfl_xor(psum, 16, 32);
    const float rinv = 1.0f / rsum;          // valid for row mrow

    // Per-accumulator-row scales: acc VGPR r holds M = r + 8*lhalf; the
    // scale for row m lives in lane m (0..15).
    float rsc[8];
#pragma unroll
    for (int r = 0; r < 8; ++r)
        rsc[r] = __shfl(rinv, r + 8 * lhalf, 32);

    // Epilogue: out[0][c][i] = vals[i][c]/rsum[i] + vars_bias[0][i][c]
#pragma unroll
    for (int cb = 0; cb < 8; ++cb) {
        const int c = cb * 16 + mrow;
#pragma unroll
        for (int r = 0; r < 8; ++r) {
            const int i = row0 + r + 8 * lhalf;
            out[(size_t)c * NN + i] =
                fmaf(acc[cb][r], rsc[r], vbias[(size_t)i * OUT_CH + c]);
        }
    }
}

// ---------------------------------------------------------------------------
extern "C" void kernel_launch(void* const* d_in, const int* in_sizes, int n_in,
                              void* d_out, int out_size, void* d_ws, size_t ws_size,
                              hipStream_t stream) {
    const float* x     = (const float*)d_in[0];   // [1,256,6144]
    const float* adj   = (const float*)d_in[1];   // [6144,6144]
    const float* W_fts = (const float*)d_in[2];   // [128,256]
    const float* w1    = (const float*)d_in[3];   // [128]
    const float* b1    = (const float*)d_in[4];   // [1]
    const float* w2    = (const float*)d_in[5];   // [128]
    const float* b2    = (const float*)d_in[6];   // [1]
    const float* vbias = (const float*)d_in[7];   // [1,6144,128]
    float* out = (float*)d_out;                   // [1,128,6144]

    char* ws = (char*)d_ws;
    float*    f  = (float*)ws;                           // 3,145,728 B
    float*    f1 = (float*)(ws + 3145728);               //    24,576 B
    float*    f2 = (float*)(ws + 3145728 + 24576);       //    24,576 B
    _Float16* Bp = (_Float16*)(ws + 3145728 + 49152);    // 1,572,864 B

    fts_kernel<<<dim3(NN / 256, OUT_CH), 256, 0, stream>>>(W_fts, x, f);
    f12_kernel<<<NN / 256, 256, 0, stream>>>(f, w1, b1, w2, b2, f1, f2);
    pack_kernel<<<(192 * 8 * 32 * 16) / 256, 256, 0, stream>>>(f, Bp);
    attn_kernel<<<NN / 16, 32, 0, stream>>>(adj, f1, f2, Bp, vbias, out);
}
